// MPNNEncoder_6433861009972
// MI455X (gfx1250) — compile-verified, matrix operand order checked by reading, not run on hardware
//
#include <hip/hip_runtime.h>
#include <hip/hip_bf16.h>

// Problem constants (match reference)
#define NN 16384      // nodes
#define EE 65536      // edges
#define GG 64         // graphs
#define PP 256        // nodes per graph
#define DD 64         // emb dim
#define HH 64         // hidden
#define DH 4096       // D*H

typedef __attribute__((ext_vector_type(16))) _Float16 v16h;
typedef __attribute__((ext_vector_type(8)))  float    v8f;

// Output flat offsets (tuple return order)
#define OFF_DX   0            // dense_x [64,256,128]
#define OFF_EF   2097152      // encoder_feature [16384,128]
#define OFF_MASK 4194304      // valid_mask [64,256] (1.0f)
#define OFF_H1   4210688      // h_1 [2,64,64]
#define OFF_H2   4218880      // h_2 [2,64,64]

// Workspace byte offsets
#define WS_AGG 0u             // float[16384*64]   4 MB
#define WS_X1  4194304u       // float[16384*64]   4 MB
#define WS_GE1 8388608u       // float[64*64]      16 KB
#define WS_PW  8404992u       // _Float16[262144]  512 KB (32B aligned)

// Single-instruction relu: v_med3_f32(x, 0, +inf)
__device__ __forceinline__ float relu1(float x) {
    return __builtin_amdgcn_fmed3f(x, 0.0f, __builtin_inff());
}

// ---------------- K0: zero the aggregation buffer ----------------
__global__ void k_zero(float* agg, int n) {
    int i = blockIdx.x * blockDim.x + threadIdx.x;
    for (; i < n; i += gridDim.x * blockDim.x) agg[i] = 0.0f;
}

// 16-bit A/B operand K-index mapping for wmma_f32_16x16x32_f16:
// half j (0..15) of a lane, laneHi = lane>=16:  K = (j&7) + 8*(2*(j>>3) + laneHi)
__device__ __forceinline__ int kmap(int j, int hi) {
    return (j & 7) + 8 * (((j >> 3) << 1) | hi);
}

// ---------------- K1: pack W_nn (f32 [64,4096]) into f16 B-operand layout ----
// packed index = ((((d*4 + h0c)*2 + o)*32 + lane)*16 + j)
__global__ void k_packW(const float* __restrict__ W_nn, _Float16* __restrict__ pw) {
    int tid = blockIdx.x * blockDim.x + threadIdx.x;   // 262144 total
    int j   = tid & 15;
    int L   = (tid >> 4) & 31;
    int o   = (tid >> 9) & 1;
    int h0c = (tid >> 10) & 3;
    int d   = tid >> 12;
    int hi  = L >> 4;
    int ln  = L & 15;
    int k   = o * 32 + kmap(j, hi);            // contraction row of W_nn
    int col = d * 64 + h0c * 16 + ln;          // column (d*H + h)
    pw[tid] = (_Float16)W_nn[k * DH + col];
}

// ---------------- K2: fused theta-GEMM + relu + per-edge matvec + scatter ----
// grid = E/32 blocks of 128 threads (4 waves). Wave w owns h-chunk [16w,16w+16).
// Each wave processes TWO 16-edge A tiles, sharing every B operand between them
// (halves the W-slab L2 traffic per WMMA).
__global__ __launch_bounds__(128) void k_msg(
    const int* __restrict__ x_tokens, const int* __restrict__ ea_tokens,
    const int* __restrict__ edge_index, const float* __restrict__ emb,
    const float* __restrict__ b_nn, const _Float16* __restrict__ pw,
    float* __restrict__ agg)
{
    __shared__ _Float16 EAs[32][64];   // edge-attr embeddings (A tiles), f16
    __shared__ float    Xs[64][32];    // x[dst] scales, TRANSPOSED: [d][edge]
    __shared__ float    Bns[4096];     // bias b_nn
    __shared__ int      Ssrc[32];      // scatter targets

    const int e0 = blockIdx.x * 32;
    const int t  = threadIdx.x;

    for (int idx = t; idx < 32 * 64; idx += 128) {
        int e = idx >> 6, k = idx & 63;
        int eg = e0 + e;
        int atok = ea_tokens[eg];
        EAs[e][k] = (_Float16)emb[atok * 64 + k];
        int dnode = edge_index[EE + eg];     // dst = edge_index[1]
        int dtok  = x_tokens[dnode];
        Xs[k][e]  = emb[dtok * 64 + k];      // transposed store
    }
    for (int idx = t; idx < 4096; idx += 128) Bns[idx] = b_nn[idx];
    if (t < 32) Ssrc[t] = edge_index[e0 + t];          // src = edge_index[0]
    __syncthreads();

    const int wave = t >> 5;            // h0 chunk 0..3
    const int lane = t & 31;
    const int hi   = lane >> 4;
    const int ln   = lane & 15;

    // Build A operands for both edge tiles (M = ln within tile)
    v16h a0, a1, a2, a3;
    #pragma unroll
    for (int j = 0; j < 16; ++j) {
        int k = kmap(j, hi);
        a0[j] = EAs[ln][k];
        a1[j] = EAs[ln][k + 32];
        a2[j] = EAs[16 + ln][k];
        a3[j] = EAs[16 + ln][k + 32];
    }

    v8f acc0 = {0.f, 0.f, 0.f, 0.f, 0.f, 0.f, 0.f, 0.f};
    v8f acc1 = {0.f, 0.f, 0.f, 0.f, 0.f, 0.f, 0.f, 0.f};

    for (int d = 0; d < 64; ++d) {
        // B operands: coalesced 32B/lane from packed W (L2 resident),
        // shared by both edge tiles.
        const v16h* bp0 = reinterpret_cast<const v16h*>(
            pw + (((size_t)(d * 4 + wave) * 2 + 0) * 32 + lane) * 16);
        const v16h* bp1 = reinterpret_cast<const v16h*>(
            pw + (((size_t)(d * 4 + wave) * 2 + 1) * 32 + lane) * 16);
        v16h b0 = *bp0;
        v16h b1 = *bp1;

        // C = 0 lowers to the inline-constant SRC2 form (no movs, no LDS dep)
        v8f c0 = {0.f, 0.f, 0.f, 0.f, 0.f, 0.f, 0.f, 0.f};
        v8f c1 = {0.f, 0.f, 0.f, 0.f, 0.f, 0.f, 0.f, 0.f};
        c0 = __builtin_amdgcn_wmma_f32_16x16x32_f16(false, a0, false, b0,
                                                    (short)0, c0, false, false);
        c1 = __builtin_amdgcn_wmma_f32_16x16x32_f16(false, a2, false, b0,
                                                    (short)0, c1, false, false);
        c0 = __builtin_amdgcn_wmma_f32_16x16x32_f16(false, a1, false, b1,
                                                    (short)0, c0, false, false);
        c1 = __builtin_amdgcn_wmma_f32_16x16x32_f16(false, a3, false, b1,
                                                    (short)0, c1, false, false);

        // bias (constant per lane for this d) + single-instr relu + scaled acc
        float bias = Bns[d * 64 + wave * 16 + ln];
        #pragma unroll
        for (int v = 0; v < 8; v += 2) {
            float2 s0 = *reinterpret_cast<const float2*>(&Xs[d][hi * 8 + v]);
            float2 s1 = *reinterpret_cast<const float2*>(&Xs[d][16 + hi * 8 + v]);
            acc0[v]     += s0.x * relu1(c0[v]     + bias);
            acc0[v + 1] += s0.y * relu1(c0[v + 1] + bias);
            acc1[v]     += s1.x * relu1(c1[v]     + bias);
            acc1[v + 1] += s1.y * relu1(c1[v + 1] + bias);
        }
    }

    // Scatter-add: C layout -> edge M = v + 8*hi (+16 for tile1), h = wave*16+ln
    int h = wave * 16 + ln;
    #pragma unroll
    for (int v = 0; v < 8; ++v) {
        int el0 = v + hi * 8;
        int el1 = 16 + el0;
        unsafeAtomicAdd(&agg[Ssrc[el0] * 64 + h], acc0[v]);
        unsafeAtomicAdd(&agg[Ssrc[el1] * 64 + h], acc1[v]);
    }
}

// ---------------- K3: x1 = agg + emb[x_tokens] @ root + conv_b ----------------
__global__ __launch_bounds__(64) void k_node(
    const int* __restrict__ x_tokens, const float* __restrict__ emb,
    const float* __restrict__ root, const float* __restrict__ conv_b,
    const float* __restrict__ agg, float* __restrict__ x1)
{
    int n = blockIdx.x;           // 16384
    int h = threadIdx.x;          // 64
    __shared__ float xr[64];
    int tok = x_tokens[n];
    xr[h] = emb[tok * 64 + h];
    __syncthreads();
    float s = 0.0f;
    #pragma unroll 8
    for (int d = 0; d < 64; ++d) s += xr[d] * root[d * 64 + h];
    x1[n * 64 + h] = agg[n * 64 + h] + s + conv_b[h];
}

// ---------------- K4: mean pool per graph ----------------
__global__ __launch_bounds__(64) void k_pool(const float* __restrict__ x1,
                                             float* __restrict__ ge1)
{
    int g = blockIdx.x;           // 64
    int h = threadIdx.x;          // 64
    float s = 0.0f;
    for (int p = 0; p < PP; ++p) s += x1[((size_t)g * PP + p) * 64 + h];
    ge1[g * 64 + h] = s * (1.0f / PP);
}

// ---------------- K5: four heads; ge = [ge1, ge1] so fold W halves -----------
__global__ __launch_bounds__(256) void k_heads(
    const float* __restrict__ ge1,
    const float* __restrict__ W1, const float* __restrict__ b1,
    const float* __restrict__ W2, const float* __restrict__ b2,
    const float* __restrict__ W3, const float* __restrict__ b3,
    const float* __restrict__ W4, const float* __restrict__ b4,
    float* __restrict__ out)
{
    int g = blockIdx.x;                  // 64
    int t = threadIdx.x;                 // 256 = 4 heads * 64 out
    __shared__ float gs[64];
    if (t < 64) gs[t] = ge1[g * 64 + t];
    __syncthreads();
    int head = t >> 6;
    int o    = t & 63;
    const float* W = (head == 0) ? W1 : (head == 1) ? W2 : (head == 2) ? W3 : W4;
    const float* b = (head == 0) ? b1 : (head == 1) ? b2 : (head == 2) ? b3 : b4;
    float s = b[o];
    #pragma unroll 8
    for (int j = 0; j < 64; ++j)
        s += gs[j] * (W[j * 64 + o] + W[(64 + j) * 64 + o]);
    // h_1 = [h1;h2], h_2 = [h3;h4]; each part is [64,64]
    int base = (head < 2) ? (OFF_H1 + (head & 1) * 4096)
                          : (OFF_H2 + (head & 1) * 4096);
    out[base + g * 64 + o] = s;
}

// ---------------- K6: dense_x, encoder_feature (fc folded), valid_mask -------
__global__ __launch_bounds__(128) void k_dense_fc(
    const float* __restrict__ x1, const float* __restrict__ W_fc,
    const float* __restrict__ b_fc, float* __restrict__ out)
{
    int n = blockIdx.x;            // 16384
    int t = threadIdx.x;           // 128
    __shared__ float xr[64];
    if (t < 64) xr[t] = x1[n * 64 + t];
    __syncthreads();
    // dense_x row = [x1_n, x1_n]
    out[OFF_DX + (size_t)n * 128 + t] = xr[t & 63];
    // encoder_feature = relu([x1,x1] @ W_fc + b): fold the two W halves
    float s = b_fc[t];
    #pragma unroll 8
    for (int d = 0; d < 64; ++d)
        s += xr[d] * (W_fc[d * 128 + t] + W_fc[(64 + d) * 128 + t]);
    out[OFF_EF + (size_t)n * 128 + t] = relu1(s);
    if (t == 0) out[OFF_MASK + n] = 1.0f;   // valid_mask all true
}

extern "C" void kernel_launch(void* const* d_in, const int* in_sizes, int n_in,
                              void* d_out, int out_size, void* d_ws, size_t ws_size,
                              hipStream_t stream) {
    const int*   x_tokens  = (const int*)d_in[0];
    const int*   ea_tokens = (const int*)d_in[1];
    const int*   edge_idx  = (const int*)d_in[2];
    // d_in[3] = batch (unused: equal-size graphs -> static layout)
    const float* emb   = (const float*)d_in[4];
    const float* W_nn  = (const float*)d_in[5];
    const float* b_nn  = (const float*)d_in[6];
    const float* root  = (const float*)d_in[7];
    const float* convb = (const float*)d_in[8];
    const float* W_fc  = (const float*)d_in[9];
    const float* b_fc  = (const float*)d_in[10];
    const float* W1 = (const float*)d_in[11]; const float* b1 = (const float*)d_in[12];
    const float* W2 = (const float*)d_in[13]; const float* b2 = (const float*)d_in[14];
    const float* W3 = (const float*)d_in[15]; const float* b3 = (const float*)d_in[16];
    const float* W4 = (const float*)d_in[17]; const float* b4 = (const float*)d_in[18];

    char* ws = (char*)d_ws;
    float*     agg = (float*)(ws + WS_AGG);
    float*     x1  = (float*)(ws + WS_X1);
    float*     ge1 = (float*)(ws + WS_GE1);
    _Float16*  pw  = (_Float16*)(ws + WS_PW);
    float*     out = (float*)d_out;

    // K0: zero scatter target
    k_zero<<<1024, 256, 0, stream>>>(agg, NN * HH);
    // K1: pack W_nn into WMMA B-operand f16 layout (512 KB, L2-resident)
    k_packW<<<262144 / 256, 256, 0, stream>>>(W_nn, pw);
    // K2: dominant fused WMMA kernel (34.4 GFLOP, theta never hits memory,
    //     B operands shared across two edge tiles)
    k_msg<<<EE / 32, 128, 0, stream>>>(x_tokens, ea_tokens, edge_idx,
                                       emb, b_nn, pw, agg);
    // K3: node update (root GEMM + bias + agg)
    k_node<<<NN, 64, 0, stream>>>(x_tokens, emb, root, convb, agg, x1);
    // K4: per-graph mean pool
    k_pool<<<GG, 64, 0, stream>>>(x1, ge1);
    // K5: four linear heads -> h_1, h_2
    k_heads<<<GG, 256, 0, stream>>>(ge1, W1, b1, W2, b2, W3, b3, W4, b4, out);
    // K6: dense_x, encoder_feature, valid_mask
    k_dense_fc<<<NN, 128, 0, stream>>>(x1, W_fc, b_fc, out);
}
